// GroupedQueryAttention_15436112462686
// MI455X (gfx1250) — matrix operand order repaired and checked by reading, not compiled, not run on hardware
//
#include <hip/hip_runtime.h>

// ---------------------------------------------------------------------------
// GQA prefill for MI455X (gfx1250, wave32).
//  * One cheap pass converts x/Wq/Wk/Wv/Wo f32 -> bf16 (v_cvt_pk_bf16_f32).
//  * All matmuls run on v_wmma_f32_16x16x32_bf16 with f32 accumulation.
//  * GEMM tiles move global->LDS via GLOBAL_LOAD_ASYNC_TO_LDS_B128 (inline
//    asm, tracked on ASYNCcnt), double-buffered so DMA overlaps the WMMAs.
//  * Attention is fused flash-style: the 2048x2048-per-head score matrix never
//    touches memory; online softmax in f32 with wave32 shfl reductions.
// ---------------------------------------------------------------------------

typedef __attribute__((ext_vector_type(16))) __bf16 v16bf;
typedef __attribute__((ext_vector_type(8)))  float v8f;

union Frag16 { v16bf v; unsigned short u[16]; uint4 q[2]; };

// f32 -> bf16 round-to-nearest-even (scalar fallback)
__device__ __forceinline__ unsigned short f2bf(float f) {
  unsigned u = __builtin_bit_cast(unsigned, f);
  u += 0x7FFFu + ((u >> 16) & 1u);
  return (unsigned short)(u >> 16);
}

// packed pair conversion: returns {bf16(lo), bf16(hi)} as one dword
__device__ __forceinline__ unsigned pk2(float lo, float hi) {
#if __has_builtin(__builtin_amdgcn_cvt_pk_bf16_f32)
  return __builtin_bit_cast(unsigned, __builtin_amdgcn_cvt_pk_bf16_f32(lo, hi));
#else
  return (unsigned)f2bf(lo) | ((unsigned)f2bf(hi) << 16);
#endif
}

// ---- async global->LDS copy (16B per lane) via inline asm -----------------
// vdst = wave-relative LDS byte offset (low 32 bits of the generic pointer),
// vaddr = 64-bit global address. Tracked on ASYNCcnt.
#if defined(__gfx1250__)
#define USE_ASYNC 1
#else
#define USE_ASYNC 0
#endif

__device__ __forceinline__ void stage16(const unsigned short* g, unsigned short* l) {
#if USE_ASYNC
  unsigned lds_off = (unsigned)(unsigned long long)l;
  unsigned long long gaddr = (unsigned long long)g;
  asm volatile("global_load_async_to_lds_b128 %0, %1, off"
               :: "v"(lds_off), "v"(gaddr) : "memory");
#else
  *(uint4*)l = *(const uint4*)g;
#endif
}

__device__ __forceinline__ void wait_async() {
#if USE_ASYNC
#if __has_builtin(__builtin_amdgcn_s_wait_asynccnt)
  __builtin_amdgcn_s_wait_asynccnt(0);
#else
  asm volatile("s_wait_asynccnt 0x0" ::: "memory");
#endif
#endif
}

// Build a 16xbf16 fragment from two contiguous 8-element LDS segments.
__device__ __forceinline__ v16bf frag_2seg(const unsigned short* s0,
                                           const unsigned short* s1) {
  Frag16 f;
#pragma unroll
  for (int i = 0; i < 8; ++i) { f.u[i] = s0[i]; f.u[8 + i] = s1[i]; }
  return f.v;
}

// Build a 16xbf16 fragment from one contiguous 16-element LDS segment.
__device__ __forceinline__ v16bf frag_16(const unsigned short* s) {
  Frag16 f;
#pragma unroll
  for (int i = 0; i < 16; ++i) f.u[i] = s[i];
  return f.v;
}

// ---------------------------------------------------------------------------
// f32 -> bf16 bulk conversion (8 elements/thread)
// ---------------------------------------------------------------------------
__global__ __launch_bounds__(256)
void cvt_f32_bf16(const float* __restrict__ src, unsigned short* __restrict__ dst,
                  int n) {
  const int i = (blockIdx.x * 256 + threadIdx.x) * 8;
  if (i >= n) return;
  float4 a = *(const float4*)(src + i);
  float4 b = *(const float4*)(src + i + 4);
  uint4 r;
  r.x = pk2(a.x, a.y); r.y = pk2(a.z, a.w);
  r.z = pk2(b.x, b.y); r.w = pk2(b.z, b.w);
  *(uint4*)(dst + i) = r;
}

// ---------------------------------------------------------------------------
// GEMM: Y[M,N] = X[M,K] @ W[N,K]^T, X/W bf16, Y bf16 or f32.
// 256 threads = 8 waves; 64x64 output tile; each wave a 16x32 sub-tile.
// K-step 64 (4 WMMAs/wave/iter); LDS double-buffered with async DMA.
// ---------------------------------------------------------------------------
#define GEMM_LSTR 72  // LDS row stride in shorts (64 data + 8 pad)

__global__ __launch_bounds__(256)
void gemm_bf16(const unsigned short* __restrict__ X,
               const unsigned short* __restrict__ W,
               void* __restrict__ Yv, int M, int N, int K, int store_bf16) {
  __shared__ unsigned short Xs[2][64 * GEMM_LSTR];
  __shared__ unsigned short Ws[2][64 * GEMM_LSTR];

  const int tid  = threadIdx.x;
  const int lane = tid & 31;
  const int wid  = tid >> 5;
  const int hi   = lane >> 4;
  const int lx   = lane & 15;

  const int mb = blockIdx.y * 64;
  const int nb = blockIdx.x * 64;
  const int wm = (wid & 3) * 16;   // 4 waves along M
  const int wn = (wid >> 2) * 32;  // 2 waves along N

  // cooperative staging: 2 chunks of 16B per thread per matrix (64x64 bf16)
  const int c0r = tid >> 3,         c0c = (tid & 7) * 8;
  const int c1r = (tid + 256) >> 3, c1c = c0c;
  const unsigned short* xg0 = X + (size_t)(mb + c0r) * K + c0c;
  const unsigned short* xg1 = X + (size_t)(mb + c1r) * K + c1c;
  const unsigned short* wg0 = W + (size_t)(nb + c0r) * K + c0c;
  const unsigned short* wg1 = W + (size_t)(nb + c1r) * K + c1c;
  unsigned short* xd[2] = {&Xs[0][c0r * GEMM_LSTR + c0c], &Xs[1][c0r * GEMM_LSTR + c0c]};
  unsigned short* xe[2] = {&Xs[0][c1r * GEMM_LSTR + c1c], &Xs[1][c1r * GEMM_LSTR + c1c]};
  unsigned short* wd[2] = {&Ws[0][c0r * GEMM_LSTR + c0c], &Ws[1][c0r * GEMM_LSTR + c0c]};
  unsigned short* we[2] = {&Ws[0][c1r * GEMM_LSTR + c1c], &Ws[1][c1r * GEMM_LSTR + c1c]};

  v8f c0 = {}, c1 = {};

  // prologue: stage tile 0 into buffer 0
  stage16(xg0, xd[0]); stage16(xg1, xe[0]);
  stage16(wg0, wd[0]); stage16(wg1, we[0]);

  for (int kk = 0; kk < K; kk += 64) {
    const int cur = (kk >> 6) & 1;
    wait_async();      // this wave's staged chunks are in LDS
    __syncthreads();   // everyone's chunks visible; prev-buffer reads done
    if (kk + 64 < K) { // stage next tile into the other buffer (overlapped)
      const int nxt = cur ^ 1;
      stage16(xg0 + kk + 64, xd[nxt]); stage16(xg1 + kk + 64, xe[nxt]);
      stage16(wg0 + kk + 64, wd[nxt]); stage16(wg1 + kk + 64, we[nxt]);
    }
#pragma unroll
    for (int ks = 0; ks < 2; ++ks) {
      const unsigned short* ap = &Xs[cur][(wm + lx) * GEMM_LSTR + ks * 32 + hi * 16];
      v16bf A  = frag_2seg(ap, ap + 8);
      v16bf B0 = frag_16(&Ws[cur][(wn + lx) * GEMM_LSTR + ks * 32 + hi * 16]);
      v16bf B1 = frag_16(&Ws[cur][(wn + 16 + lx) * GEMM_LSTR + ks * 32 + hi * 16]);
      c0 = __builtin_amdgcn_wmma_f32_16x16x32_bf16(false, A, false, B0,
                                                   (short)0, c0, false, false);
      c1 = __builtin_amdgcn_wmma_f32_16x16x32_bf16(false, A, false, B1,
                                                   (short)0, c1, false, false);
    }
  }

  // C layout: col = lx (+16), VGPR r -> row r + 8*hi
  if (store_bf16) {
    unsigned short* Y = (unsigned short*)Yv;
#pragma unroll
    for (int r = 0; r < 8; ++r) {
      const size_t row = (size_t)(mb + wm + r + 8 * hi);
      Y[row * N + nb + wn + lx]      = f2bf(c0[r]);
      Y[row * N + nb + wn + lx + 16] = f2bf(c1[r]);
    }
  } else {
    float* Y = (float*)Yv;
#pragma unroll
    for (int r = 0; r < 8; ++r) {
      const size_t row = (size_t)(mb + wm + r + 8 * hi);
      Y[row * N + nb + wn + lx]      = c0[r];
      Y[row * N + nb + wn + lx + 16] = c1[r];
    }
  }
}

// ---------------------------------------------------------------------------
// Fused causal flash attention (grouped KV), bf16 Q/K/V/ctx.
// Grid: (S/64, H, B); block = 128 threads = 4 waves; wave owns 16 q rows.
// ---------------------------------------------------------------------------
#define AHD 64
#define AGS 4
#define K_LSTR 72  // Ks: [key][feat], 64 + 8 pad
#define V_LSTR 40  // Vs: [feat][key], 32 + 8 pad (transposed for P@V B-frags)
#define P_LSTR 40  // Ps: [row][key],  32 + 8 pad

__global__ __launch_bounds__(128)
void gqa_flash_attn(const unsigned short* __restrict__ Qb,
                    const unsigned short* __restrict__ Kb,
                    const unsigned short* __restrict__ Vb,
                    unsigned short* __restrict__ Cb,
                    int B, int S, int H, int G) {
  __shared__ unsigned short Ks[32 * K_LSTR];
  __shared__ unsigned short Vs[64 * V_LSTR];
  __shared__ unsigned short Ps[4][16 * P_LSTR];

  const int tid  = threadIdx.x;
  const int lane = tid & 31;
  const int wid  = tid >> 5;
  const int hi   = lane >> 4;
  const int lx   = lane & 15;

  const int h  = blockIdx.y;
  const int g  = h / AGS;
  const int b  = blockIdx.z;
  const int qb = blockIdx.x * 64;
  const int qw = qb + wid * 16;
  const int D  = H * AHD;
  const float scale = 0.125f;  // 1/sqrt(64), applied to f32 scores (exact)

  // Q fragments (A-matrix 16x64 = two 16x32 frags): straight b128 loads
  v16bf aq[2];
  {
    const unsigned short* qrow = Qb + ((size_t)(b * S + qw + lx)) * D + h * AHD;
#pragma unroll
    for (int kk = 0; kk < 2; ++kk) {
      const unsigned short* p0 = qrow + kk * 32 + hi * 8;
      Frag16 f;
      f.q[0] = *(const uint4*)p0;         // K segs {0..7}/{8..15}
      f.q[1] = *(const uint4*)(p0 + 16);  // K segs {16..23}/{24..31}
      aq[kk] = f.v;
    }
  }

  float mrun[8], lrun[8];
  v8f o[4] = {{}, {}, {}, {}};
#pragma unroll
  for (int r = 0; r < 8; ++r) { mrun[r] = -__builtin_inff(); lrun[r] = 0.f; }

  const int kend = qb + 64;
  for (int kb = 0; kb < kend; kb += 32) {
    // ---- cooperative K/V chunk staging (always, all 128 threads) ----
    const int key = tid & 31;
    const int fb  = (tid >> 5) * 16;  // 0,16,32,48
    const size_t rbase = ((size_t)(b * S + kb + key) * G + g) * AHD + fb;
    __syncthreads();  // previous chunk's fragment reads complete
    stage16(Kb + rbase,     &Ks[key * K_LSTR + fb]);      // async DMA
    stage16(Kb + rbase + 8, &Ks[key * K_LSTR + fb + 8]);
    {  // V transposed into LDS (per-element scatter; layout change)
      Frag16 vf;
      vf.q[0] = *(const uint4*)(Vb + rbase);
      vf.q[1] = *(const uint4*)(Vb + rbase + 8);
#pragma unroll
      for (int i = 0; i < 16; ++i) Vs[(fb + i) * V_LSTR + key] = vf.u[i];
    }
    wait_async();
    __syncthreads();

    if (kb <= qw + 15) {  // causal: this wave needs this chunk
      // scores S = Q @ K^T : two 16x16 tiles, K-dim 64 in 2 steps
      v8f s[2];
#pragma unroll
      for (int t = 0; t < 2; ++t) {
        v8f acc = {};
#pragma unroll
        for (int kk = 0; kk < 2; ++kk) {
          v16bf Bk = frag_2seg(&Ks[(t * 16 + lx) * K_LSTR + kk * 32 + hi * 8], &Ks[(t * 16 + lx) * K_LSTR + kk * 32 + hi * 8 + 16]);
          acc = __builtin_amdgcn_wmma_f32_16x16x32_bf16(false, aq[kk], false, Bk,
                                                        (short)0, acc, false, false);
        }
        s[t] = acc;
      }

      // scale + causal mask + row max (row = r + 8*hi, col group = lx)
      float mc[8];
#pragma unroll
      for (int r = 0; r < 8; ++r) {
        const int qpos = qw + r + 8 * hi;
        float s0 = s[0][r] * scale, s1 = s[1][r] * scale;
        if (kb + lx > qpos)      s0 = -3.0e38f;
        if (kb + 16 + lx > qpos) s1 = -3.0e38f;
        s[0][r] = s0; s[1][r] = s1;
        mc[r] = fmaxf(s0, s1);
      }
#pragma unroll
      for (int off = 1; off < 16; off <<= 1)
#pragma unroll
        for (int r = 0; r < 8; ++r)
          mc[r] = fmaxf(mc[r], __shfl_xor(mc[r], off, 16));

      float alpha[8], lc[8];
#pragma unroll
      for (int r = 0; r < 8; ++r) {
        const float mn = fmaxf(mrun[r], mc[r]);
        alpha[r] = __expf(mrun[r] - mn);  // exp(-inf)=0 on first chunk
        mrun[r] = mn;
        const float p0 = __expf(s[0][r] - mn);
        const float p1 = __expf(s[1][r] - mn);
        s[0][r] = p0; s[1][r] = p1;
        lc[r] = p0 + p1;
      }
#pragma unroll
      for (int off = 1; off < 16; off <<= 1)
#pragma unroll
        for (int r = 0; r < 8; ++r)
          lc[r] += __shfl_xor(lc[r], off, 16);
#pragma unroll
      for (int r = 0; r < 8; ++r) lrun[r] = lrun[r] * alpha[r] + lc[r];
#pragma unroll
      for (int j = 0; j < 4; ++j)
#pragma unroll
        for (int r = 0; r < 8; ++r) o[j][r] *= alpha[r];

      // P: C-layout -> bf16 A-layout via per-wave LDS round-trip
      unsigned short* pw = &Ps[wid][0];
#pragma unroll
      for (int r = 0; r < 8; ++r) {
        const int row = r + 8 * hi;
        pw[row * P_LSTR + lx]      = f2bf(s[0][r]);
        pw[row * P_LSTR + 16 + lx] = f2bf(s[1][r]);
      }
      __builtin_amdgcn_wave_barrier();   // LDS is in-order per wave; just
      asm volatile("" ::: "memory");     // block compiler reordering

      const unsigned short* pr = &pw[lx * P_LSTR + hi * 8];
      v16bf Ap = frag_2seg(pr, pr + 16);

      // O += P @ V : four 16x16 n-tiles, K-dim 32
#pragma unroll
      for (int j = 0; j < 4; ++j) {
        v16bf Bv = frag_2seg(&Vs[(j * 16 + lx) * V_LSTR + hi * 8], &Vs[(j * 16 + lx) * V_LSTR + hi * 8 + 16]);
        o[j] = __builtin_amdgcn_wmma_f32_16x16x32_bf16(false, Ap, false, Bv,
                                                       (short)0, o[j], false, false);
      }
    }
  }

  // normalize and store bf16 ctx (b, s, h*64 + d)
#pragma unroll
  for (int r = 0; r < 8; ++r) {
    const float inv = 1.0f / lrun[r];
    const size_t row = (size_t)(b * S + qw + r + 8 * hi);
    unsigned short* crow = Cb + row * D + h * AHD + lx;
#pragma unroll
    for (int j = 0; j < 4; ++j) crow[j * 16] = f2bf(o[j][r] * inv);
  }
}

// ---------------------------------------------------------------------------
// Launch
// ---------------------------------------------------------------------------
extern "C" void kernel_launch(void* const* d_in, const int* in_sizes, int n_in,
                              void* d_out, int out_size, void* d_ws, size_t ws_size,
                              hipStream_t stream) {
  (void)in_sizes; (void)n_in; (void)out_size; (void)ws_size;
  constexpr int B = 2, S = 2048, D = 2048, H = 32, G = 8;
  constexpr int MS = B * S;        // 4096
  constexpr int KV = G * (D / H);  // 512

  const float* x  = (const float*)d_in[0];
  const float* Wq = (const float*)d_in[1];
  const float* Wk = (const float*)d_in[2];
  const float* Wv = (const float*)d_in[3];
  const float* Wo = (const float*)d_in[4];
  float* out = (float*)d_out;

  unsigned short* w16 = (unsigned short*)d_ws;
  size_t off = 0;
  auto alloc = [&](size_t n) { unsigned short* p = w16 + off; off += n; return p; };
  unsigned short* xb  = alloc((size_t)MS * D);
  unsigned short* Wqb = alloc((size_t)D * D);
  unsigned short* Wkb = alloc((size_t)KV * D);
  unsigned short* Wvb = alloc((size_t)KV * D);
  unsigned short* Wob = alloc((size_t)D * D);
  unsigned short* Qb  = alloc((size_t)MS * D);
  unsigned short* Kb  = alloc((size_t)MS * KV);
  unsigned short* Vb  = alloc((size_t)MS * KV);
  unsigned short* Cb  = alloc((size_t)MS * D);

  // one-time bf16 conversions (bandwidth-bound, ~5us total)
  cvt_f32_bf16<<<(MS * D) / 2048, 256, 0, stream>>>(x,  xb,  MS * D);
  cvt_f32_bf16<<<(D * D) / 2048,  256, 0, stream>>>(Wq, Wqb, D * D);
  cvt_f32_bf16<<<(KV * D) / 2048, 256, 0, stream>>>(Wk, Wkb, KV * D);
  cvt_f32_bf16<<<(KV * D) / 2048, 256, 0, stream>>>(Wv, Wvb, KV * D);
  cvt_f32_bf16<<<(D * D) / 2048,  256, 0, stream>>>(Wo, Wob, D * D);

  // projections: Y = X @ W^T  (bf16 out)
  gemm_bf16<<<dim3(D / 64, MS / 64), 256, 0, stream>>>(xb, Wqb, Qb, MS, D, D, 1);
  gemm_bf16<<<dim3(KV / 64, MS / 64), 256, 0, stream>>>(xb, Wkb, Kb, MS, KV, D, 1);
  gemm_bf16<<<dim3(KV / 64, MS / 64), 256, 0, stream>>>(xb, Wvb, Vb, MS, KV, D, 1);

  // fused causal grouped attention -> bf16 ctx
  gqa_flash_attn<<<dim3(S / 64, H, B), 128, 0, stream>>>(Qb, Kb, Vb, Cb, B, S, H, G);

  // output projection (f32 out)
  gemm_bf16<<<dim3(D / 64, MS / 64), 256, 0, stream>>>(Cb, Wob, out, MS, D, D, 0);
}
